// IPEXEmptyINT4Linear_56659208569010
// MI455X (gfx1250) — matrix operand order repaired and checked by reading, not compiled, not run on hardware
//
#include <hip/hip_runtime.h>

typedef __attribute__((ext_vector_type(16))) _Float16 v16h;
typedef __attribute__((ext_vector_type(8)))  _Float16 v8h;
typedef __attribute__((ext_vector_type(2)))  _Float16 v2h;
typedef __attribute__((ext_vector_type(8)))  float    v8f;
typedef __attribute__((ext_vector_type(4)))  float    v4f;
typedef __attribute__((ext_vector_type(2)))  float    v2f;
typedef __attribute__((ext_vector_type(2)))  int      v2i;
typedef __attribute__((ext_vector_type(4)))  unsigned int v4u;

#define M_DIM 4096
#define K_DIM 4096
#define N_DIM 11008
#define BM 128
#define BN 128
#define BK 32
#define NKB (K_DIM / BK)   // 128 K-steps
#define LDA 40             // padded LDS row stride (halves): 80B rows, 16B-aligned chunks

__global__ __launch_bounds__(256)
void int4_gemm_wmma_f16(const float* __restrict__ A,
                        const int*   __restrict__ QW,
                        const float* __restrict__ SC,
                        const int*   __restrict__ ZP,
                        const float* __restrict__ BIAS,
                        float* __restrict__ OUT)
{
    __shared__ _Float16 As[2][BM][LDA];   // activations, m-major
    __shared__ _Float16 Ws[2][BN][LDA];   // dequantized weights, n-major (k inner)

    const int tid  = threadIdx.x;
    const int wave = tid >> 5;
    const int lane = tid & 31;
    const int hlf  = lane >> 4;
    const int l16  = lane & 15;

    const int nb = blockIdx.x;           // N tile
    const int mb = blockIdx.y;           // M tile

    const int wm = (wave >> 2) * 64;     // wave M offset in block tile
    const int wn = (wave & 3) * 32;      // wave N offset in block tile

    // A staging: 2 passes; thread owns row p*64 + arow, cols acol..acol+7
    const int arow = tid >> 2;           // 0..63
    const int acol = (tid & 3) * 8;      // 0,8,16,24

    // W staging: thread owns columns wn0, wn0+1 and k rows wk0..wk0+7
    const int wn0 = (tid & 63) * 2;      // 0..126
    const int wk0 = (tid >> 6) * 8;      // 0,8,16,24

    v4f a_reg[2][2];
    v2i q_reg[8];
    v2f sc_reg;
    v2i zp_reg;

    v8f acc[4][2];
#pragma unroll
    for (int i = 0; i < 4; ++i)
#pragma unroll
        for (int j = 0; j < 2; ++j)
            acc[i][j] = (v8f){0.f, 0.f, 0.f, 0.f, 0.f, 0.f, 0.f, 0.f};

    auto load_global = [&](int kb) {
#pragma unroll
        for (int p = 0; p < 2; ++p) {
            const float* ap = A + (size_t)(mb * BM + p * 64 + arow) * K_DIM
                                + (size_t)kb * BK + acol;
            a_reg[p][0] = *(const v4f*)(ap);
            a_reg[p][1] = *(const v4f*)(ap + 4);
        }
        // qweight is streamed once -> non-temporal, keep A resident in L2
#pragma unroll
        for (int i = 0; i < 8; ++i)
            q_reg[i] = __builtin_nontemporal_load(
                (const v2i*)(QW + (size_t)(kb * BK + wk0 + i) * N_DIM
                                + (size_t)nb * BN + wn0));
        const int g = kb >> 2;           // BK=32, G=128 -> group constant per K-step
        sc_reg = *(const v2f*)(SC + (size_t)g * N_DIM + (size_t)nb * BN + wn0);
        zp_reg = *(const v2i*)(ZP + (size_t)g * N_DIM + (size_t)nb * BN + wn0);
    };

    auto cvt_store = [&](int buf) {
        // activations: fp32 -> f16, one b128 store per pass
#pragma unroll
        for (int p = 0; p < 2; ++p) {
            v8h h;
#pragma unroll
            for (int i = 0; i < 4; ++i) {
                h[i]     = (_Float16)a_reg[p][0][i];
                h[4 + i] = (_Float16)a_reg[p][1][i];
            }
            *(v8h*)&As[buf][p * 64 + arow][acol] = h;
        }
        // weights: (q - zp) * scale via the 0x6400 exponent-bias trick:
        // f16bits(0x6400|v) == 1024+v exactly for 0<=v<16, so the bias cancels
        // in the packed subtract. All math stays in v_pk_* f16 ops.
#pragma unroll
        for (int s = 0; s < 2; ++s) {
            const unsigned zb = ((unsigned)zp_reg[s] | 0x6400u) * 0x00010001u; // splat
            const v2h zh = __builtin_bit_cast(v2h, zb);
            const _Float16 sh = (_Float16)sc_reg[s];
            const v2h sc2 = {sh, sh};
            v4u wbits;
#pragma unroll
            for (int j = 0; j < 4; ++j) {
                const unsigned qb = ((unsigned)q_reg[2 * j][s]
                                  | ((unsigned)q_reg[2 * j + 1][s] << 16))
                                  | 0x64006400u;
                const v2h qh = __builtin_bit_cast(v2h, qb);
                const v2h w  = (qh - zh) * sc2;
                wbits[j] = __builtin_bit_cast(unsigned, w);
            }
            *(v4u*)&Ws[buf][wn0 + s][wk0] = wbits;   // 8 f16, one b128 store
        }
    };

    auto compute = [&](int buf) {
        // B fragments: 32x16, lane holds column n = l16, K chunks [8*hlf,+8) and [16+8*hlf,+8)
        v16h bfrag[2];
#pragma unroll
        for (int nt = 0; nt < 2; ++nt) {
            const _Float16* bp = &Ws[buf][wn + nt * 16 + l16][hlf * 8];
            v8h b0 = *(const v8h*)(bp);
            v8h b1 = *(const v8h*)(bp + 16);
            bfrag[nt] = __builtin_shufflevector(b0, b1,
                0, 1, 2, 3, 4, 5, 6, 7, 8, 9, 10, 11, 12, 13, 14, 15);
        }
        // A fragments: 16x32, lane row m = l16, K chunks [8*hlf,+8) and [16+8*hlf,+8)
#pragma unroll
        for (int mt = 0; mt < 4; ++mt) {
            const _Float16* ap = &As[buf][wm + mt * 16 + l16][hlf * 8];
            v8h a0 = *(const v8h*)(ap);
            v8h a1 = *(const v8h*)(ap + 16);
            v16h af = __builtin_shufflevector(a0, a1,
                0, 1, 2, 3, 4, 5, 6, 7, 8, 9, 10, 11, 12, 13, 14, 15);
#pragma unroll
            for (int nt = 0; nt < 2; ++nt)
                acc[mt][nt] = __builtin_amdgcn_wmma_f32_16x16x32_f16(
                    false, af, false, bfrag[nt], (short)0, acc[mt][nt], false, false);
        }
    };

    // ---- pipelined main loop: one barrier per K-step, LDS ping-pong ----
    load_global(0);
    cvt_store(0);
    __syncthreads();
    for (int kb = 0; kb < NKB; ++kb) {
        const int cur = kb & 1;
        if (kb + 1 < NKB) load_global(kb + 1);   // prefetch next tile to VGPRs
        compute(cur);                             // 8x v_wmma per iteration
        if (kb + 1 < NKB) cvt_store(cur ^ 1);     // fill other buffer
        __syncthreads();
    }

    // ---- epilogue: C/D layout VGPR v -> row = 8*hlf + v, col = l16 ----
    // output is streamed once -> non-temporal stores, keep A/L2 warm
#pragma unroll
    for (int nt = 0; nt < 2; ++nt) {
        const int col = nb * BN + wn + nt * 16 + l16;
        const float bv = BIAS[col];
#pragma unroll
        for (int mt = 0; mt < 4; ++mt) {
            const int mrow = mb * BM + wm + mt * 16 + hlf * 8;
#pragma unroll
            for (int v = 0; v < 8; ++v)
                __builtin_nontemporal_store(acc[mt][nt][v] + bv,
                    OUT + (size_t)(mrow + v) * N_DIM + col);
        }
    }
}

extern "C" void kernel_launch(void* const* d_in, const int* in_sizes, int n_in,
                              void* d_out, int out_size, void* d_ws, size_t ws_size,
                              hipStream_t stream) {
    const float* A  = (const float*)d_in[0];   // input  [M,K] fp32
    const int*   QW = (const int*)  d_in[1];   // qweight [K,N] int32 (0..15)
    const float* SC = (const float*)d_in[2];   // scales [K/G,N] fp32
    const int*   ZP = (const int*)  d_in[3];   // qzeros [K/G,N] int32
    const float* BI = (const float*)d_in[4];   // bias [N] fp32
    // d_in[5] = group_size scalar (=128), baked into kernel constants

    dim3 grid(N_DIM / BN, M_DIM / BM);         // 86 x 32
    dim3 block(256);                           // 8 wave32s
    hipLaunchKernelGGL(int4_gemm_wmma_f16, grid, block, 0, stream,
                       A, QW, SC, ZP, BI, (float*)d_out);
}
